// SpiralClassifier_10471130268093
// MI455X (gfx1250) — matrix operand order repaired and checked by reading, not compiled, not run on hardware
//
#include <hip/hip_runtime.h>
#include <hip/hip_bf16.h>

typedef __bf16 v16bf __attribute__((ext_vector_type(16)));
typedef float  v8f   __attribute__((ext_vector_type(8)));
typedef unsigned int v8u __attribute__((ext_vector_type(8)));

constexpr int BATCH = 32;
constexpr int SEQ   = 9;

__device__ __forceinline__ unsigned short f2bf(float f) {
  unsigned int u = __float_as_uint(f);
  u += 0x7fffu + ((u >> 16) & 1u);   // round-to-nearest-even
  return (unsigned short)(u >> 16);
}

// ---------------------------------------------------------------------------
// Fused gather + GEMM (bf16 WMMA, f32 accum) + bias + ELU.
// One workgroup (8 waves) computes a 16(M) x N_OUT tile.
// ---------------------------------------------------------------------------
template<int C_IN, int N_IN, int N_OUT, int TPW, int K_PAD, int K_REAL>
__global__ __launch_bounds__(256, 4)
void spiral_conv_wmma(const float* __restrict__ hin,
                      const int* __restrict__ spiral,
                      const unsigned short* __restrict__ wbf,   // [N_OUT][K_PAD] bf16
                      const float* __restrict__ bias,
                      float* __restrict__ out)                  // [B*N_IN][N_OUT] f32
{
  constexpr int KC  = (K_PAD >= 64) ? 64 : 32;   // K chunk staged in LDS
  constexpr int LDP = KC + 8;                    // 144B row stride: 16B-aligned rows,
                                                 // 36*r mod 64 banks = permutation
  __shared__ unsigned short atile[16 * LDP];

  const int tid   = threadIdx.x;
  const int lane  = tid & 31;
  const int wave  = tid >> 5;
  const int lhalf = lane >> 4;   // 0: lanes 0-15, 1: lanes 16-31
  const int l16   = lane & 15;

  const int m_base = blockIdx.x * 16;
  const int b      = m_base / N_IN;   // N_IN % 16 == 0 -> constant per tile
  const int n_base = m_base % N_IN;

  v8f acc[TPW];
  const v8f vzero = {0.f, 0.f, 0.f, 0.f, 0.f, 0.f, 0.f, 0.f};
#pragma unroll
  for (int t = 0; t < TPW; ++t) acc[t] = vzero;

  for (int kc = 0; kc < K_PAD; kc += KC) {
    // -------- cooperative gather of 16 x KC A-chunk into LDS (f32 -> bf16)
    if constexpr ((C_IN % 4) == 0 && KC == 64) {
      // 4 consecutive k share one spiral entry and 4 contiguous 16B-aligned f32:
      // 1 spiral load + 1 global_load_b128 + 2 packs + 1 ds_store_b64 per thread.
      constexpr int QPR = KC / 4;                 // quads per row (16)
      static_assert(16 * QPR == 256, "one quad per thread");
      const int r  = tid / QPR;
      const int kk = (tid % QPR) * 4;
      const int k  = kc + kk;
      const int s  = k / C_IN;
      const int c  = k - s * C_IN;                // c+3 < C_IN since C_IN % 4 == 0
      const int src_n = spiral[(n_base + r) * SEQ + s];
      const float4 f4 = *reinterpret_cast<const float4*>(
          &hin[((size_t)b * N_IN + src_n) * C_IN + c]);
      uint2 pk;
      pk.x = (unsigned int)f2bf(f4.x) | ((unsigned int)f2bf(f4.y) << 16);
      pk.y = (unsigned int)f2bf(f4.z) | ((unsigned int)f2bf(f4.w) << 16);
      *reinterpret_cast<uint2*>(&atile[r * LDP + kk]) = pk;
    } else {
#pragma unroll
      for (int e = tid; e < 16 * KC; e += 256) {
        const int r  = e / KC;
        const int kk = e - r * KC;
        const int k  = kc + kk;
        float f = 0.f;
        if (k < K_REAL) {
          const int s     = k / C_IN;
          const int c     = k - s * C_IN;
          const int src_n = spiral[(n_base + r) * SEQ + s];
          f = hin[((size_t)b * N_IN + src_n) * C_IN + c];
        }
        atile[r * LDP + kk] = f2bf(f);
      }
    }
    __syncthreads();

#pragma unroll
    for (int ks = 0; ks < KC; ks += 32) {
      // -------- A fragment (16x32 bf16): lane = M; two 16B-aligned ds_load_b128
      const uint4* ap = reinterpret_cast<const uint4*>(
          &atile[l16 * LDP + ks + lhalf * 8]);
      const uint4 a0 = ap[0];   // v0..3 : K pairs {0,2,4,6}+lhalf*8
      const uint4 a1 = ap[2];   // v4..7 : +16 halfwords (K base +16)
      v8u au;
      au[0] = a0.x; au[1] = a0.y; au[2] = a0.z; au[3] = a0.w;
      au[4] = a1.x; au[5] = a1.y; au[6] = a1.z; au[7] = a1.w;
      const v16bf afrag = __builtin_bit_cast(v16bf, au);

#pragma unroll
      for (int t = 0; t < TPW; ++t) {
        // -------- B fragment (32x16 bf16): lane reads K {0..7,16..23}+lhalf*8 of one W row
        const int ncol = (wave * TPW + t) * 16 + l16;
        const uint4* wp = reinterpret_cast<const uint4*>(
            wbf + (size_t)ncol * K_PAD + (kc + ks + lhalf * 8));
        const uint4 b0v = wp[0];
        const uint4 b1v = wp[2];
        v8u bu;
        bu[0] = b0v.x; bu[1] = b0v.y; bu[2] = b0v.z; bu[3] = b0v.w;
        bu[4] = b1v.x; bu[5] = b1v.y; bu[6] = b1v.z; bu[7] = b1v.w;
        const v16bf bfrag = __builtin_bit_cast(v16bf, bu);

        acc[t] = __builtin_amdgcn_wmma_f32_16x16x32_bf16(
            false, afrag, false, bfrag, (short)0, acc[t], false, false);
      }
    }
    __syncthreads();
  }

  // -------- epilogue: bias + ELU; each store covers two full 64B lines
#pragma unroll
  for (int t = 0; t < TPW; ++t) {
    const int   ncol = (wave * TPW + t) * 16 + l16;
    const float bv   = bias[ncol];
#pragma unroll
    for (int v = 0; v < 8; ++v) {
      const int m = m_base + lhalf * 8 + v;   // C/D layout: M = (lane/16)*8 + v
      float x = acc[t][v] + bv;
      x = (x > 0.f) ? x : (__expf(x) - 1.f);
      out[(size_t)m * N_OUT + ncol] = x;
    }
  }
}

// ---------------------------------------------------------------------------
// COO pool: out[b, rows[i], c] += vals[i] * conv[b, cols[i], c]
// ---------------------------------------------------------------------------
template<int N_IN, int N_OUT, int C>
__global__ void pool_scatter(const float* __restrict__ conv,
                             const int* __restrict__ rows,
                             const int* __restrict__ cols,
                             const float* __restrict__ vals,
                             float* __restrict__ outp)
{
  const size_t tid   = (size_t)blockIdx.x * blockDim.x + threadIdx.x;
  const size_t total = (size_t)N_IN * BATCH * C;
  if (tid >= total) return;
  const int    c   = (int)(tid % C);
  const size_t rem = tid / C;
  const int    bb  = (int)(rem % BATCH);
  const int    i   = (int)(rem / BATCH);
  const float  v   = vals[i] * conv[((size_t)bb * N_IN + cols[i]) * C + c];
  atomicAdd(&outp[((size_t)bb * N_OUT + rows[i]) * C + c], v);
}

__global__ void zero_f32x4(float4* __restrict__ p, size_t n4) {
  size_t i      = (size_t)blockIdx.x * blockDim.x + threadIdx.x;
  size_t stride = (size_t)gridDim.x * blockDim.x;
  const float4 z = {0.f, 0.f, 0.f, 0.f};
  for (; i < n4; i += stride) p[i] = z;
}

__global__ void convert_w(const float* __restrict__ src, unsigned short* __restrict__ dst,
                          int n_out, int k_src, int k_pad) {
  const int t = blockIdx.x * blockDim.x + threadIdx.x;
  if (t >= n_out * k_pad) return;
  const int o = t / k_pad, k = t - o * k_pad;
  const float f = (k < k_src) ? src[o * k_src + k] : 0.f;
  dst[t] = f2bf(f);
}

__global__ void mean_pool(const float* __restrict__ h, float* __restrict__ feat) {
  const int t = blockIdx.x * blockDim.x + threadIdx.x;   // BATCH*512
  if (t >= BATCH * 512) return;
  const int bb = t / 512, c = t - bb * 512;
  float s = 0.f;
  for (int n = 0; n < 256; ++n) s += h[((size_t)bb * 256 + n) * 512 + c];
  feat[t] = s * (1.f / 256.f);
}

__global__ void fc_layer(const float* __restrict__ in, const float* __restrict__ W,
                         const float* __restrict__ bias, float* __restrict__ out,
                         int n_out, int k, int do_relu) {
  const int t = blockIdx.x * blockDim.x + threadIdx.x;
  if (t >= BATCH * n_out) return;
  const int bb = t / n_out, o = t - bb * n_out;
  float s = bias[o];
  for (int kk = 0; kk < k; ++kk) s += in[bb * k + kk] * W[o * k + kk];
  if (do_relu) s = fmaxf(s, 0.f);
  out[t] = s;
}

// ---------------------------------------------------------------------------
extern "C" void kernel_launch(void* const* d_in, const int* in_sizes, int n_in,
                              void* d_out, int out_size, void* d_ws, size_t ws_size,
                              hipStream_t stream)
{
  const float* x       = (const float*)d_in[0];
  const int*   spiral0 = (const int*)d_in[1];
  const int*   spiral1 = (const int*)d_in[2];
  const int*   spiral2 = (const int*)d_in[3];
  const int*   d0_rows = (const int*)d_in[4];
  const int*   d0_cols = (const int*)d_in[5];
  const float* d0_vals = (const float*)d_in[6];
  const int*   d1_rows = (const int*)d_in[7];
  const int*   d1_cols = (const int*)d_in[8];
  const float* d1_vals = (const float*)d_in[9];
  const int*   d2_rows = (const int*)d_in[10];
  const int*   d2_cols = (const int*)d_in[11];
  const float* d2_vals = (const float*)d_in[12];
  const float* W0  = (const float*)d_in[13];
  const float* b0  = (const float*)d_in[14];
  const float* W1  = (const float*)d_in[15];
  const float* b1  = (const float*)d_in[16];
  const float* W2  = (const float*)d_in[17];
  const float* b2  = (const float*)d_in[18];
  const float* Wc1 = (const float*)d_in[19];
  const float* bc1 = (const float*)d_in[20];
  const float* Wc2 = (const float*)d_in[21];
  const float* bc2 = (const float*)d_in[22];

  // --- workspace bump allocator (256B aligned) ---
  char*  ws  = (char*)d_ws;
  size_t off = 0;
  auto alloc = [&](size_t bytes) -> void* {
    void* p = ws + off;
    off = (off + bytes + 255) & ~(size_t)255;
    return p;
  };
  unsigned short* w0bf = (unsigned short*)alloc((size_t)128 * 32   * 2);
  unsigned short* w1bf = (unsigned short*)alloc((size_t)256 * 1152 * 2);
  unsigned short* w2bf = (unsigned short*)alloc((size_t)512 * 2304 * 2);
  float* conv  = (float*)alloc((size_t)BATCH * 16384 * 128 * 4);  // reused all layers
  float* p1    = (float*)alloc((size_t)BATCH * 4096  * 128 * 4);
  float* p2    = (float*)alloc((size_t)BATCH * 1024  * 256 * 4);
  float* p3    = (float*)alloc((size_t)BATCH * 256   * 512 * 4);
  float* feat  = (float*)alloc((size_t)BATCH * 512 * 4);
  float* fcbuf = (float*)alloc((size_t)BATCH * 512 * 4);
  (void)ws_size; (void)in_sizes; (void)n_in; (void)out_size;

  // --- weights -> bf16 (W0 padded K 27 -> 32) ---
  convert_w<<<(128 * 32   + 255) / 256, 256, 0, stream>>>(W0, w0bf, 128, 27,   32);
  convert_w<<<(256 * 1152 + 255) / 256, 256, 0, stream>>>(W1, w1bf, 256, 1152, 1152);
  convert_w<<<(512 * 2304 + 255) / 256, 256, 0, stream>>>(W2, w2bf, 512, 2304, 2304);

  // --- zero pooled accumulators (required every replay) ---
  zero_f32x4<<<2048, 256, 0, stream>>>((float4*)p1, (size_t)BATCH * 4096 * 128 / 4);
  zero_f32x4<<<2048, 256, 0, stream>>>((float4*)p2, (size_t)BATCH * 1024 * 256 / 4);
  zero_f32x4<<<2048, 256, 0, stream>>>((float4*)p3, (size_t)BATCH * 256  * 512 / 4);

  // --- layer 0: (B,16384,3) --gather9--> K=27 GEMM -> 128, ELU, pool -> (B,4096,128)
  spiral_conv_wmma<3, 16384, 128, 1, 32, 27>
      <<<BATCH * 16384 / 16, 256, 0, stream>>>(x, spiral0, w0bf, b0, conv);
  pool_scatter<16384, 4096, 128>
      <<<(unsigned)(((size_t)16384 * BATCH * 128 + 255) / 256), 256, 0, stream>>>(
          conv, d0_rows, d0_cols, d0_vals, p1);

  // --- layer 1: K=1152 GEMM -> 256, ELU, pool -> (B,1024,256)
  spiral_conv_wmma<128, 4096, 256, 2, 1152, 1152>
      <<<BATCH * 4096 / 16, 256, 0, stream>>>(p1, spiral1, w1bf, b1, conv);
  pool_scatter<4096, 1024, 256>
      <<<(unsigned)(((size_t)4096 * BATCH * 256 + 255) / 256), 256, 0, stream>>>(
          conv, d1_rows, d1_cols, d1_vals, p2);

  // --- layer 2: K=2304 GEMM -> 512, ELU, pool -> (B,256,512)
  spiral_conv_wmma<256, 1024, 512, 4, 2304, 2304>
      <<<BATCH * 1024 / 16, 256, 0, stream>>>(p2, spiral2, w2bf, b2, conv);
  pool_scatter<1024, 256, 512>
      <<<(unsigned)(((size_t)1024 * BATCH * 512 + 255) / 256), 256, 0, stream>>>(
          conv, d2_rows, d2_cols, d2_vals, p3);

  // --- head: mean over N3, FC 512->512 relu, FC 512->10
  mean_pool<<<(BATCH * 512 + 255) / 256, 256, 0, stream>>>(p3, feat);
  fc_layer<<<(BATCH * 512 + 255) / 256, 256, 0, stream>>>(feat, Wc1, bc1, fcbuf, 512, 512, 1);
  fc_layer<<<(BATCH * 10  + 255) / 256, 256, 0, stream>>>(fcbuf, Wc2, bc2, (float*)d_out, 10, 512, 0);
}
